// MultiViewGNN_37993280700423
// MI455X (gfx1250) — matrix operand order repaired and checked by reading, not compiled, not run on hardware
//
#include <hip/hip_runtime.h>

typedef _Float16 f16_t;
typedef __attribute__((ext_vector_type(16))) _Float16 v16h;
typedef __attribute__((ext_vector_type(8)))  float    v8f;

#define LDA_S 40   // LDS stride (halfs) for A tile rows (128x32)
#define LDB_S 40   // LDS stride (halfs) for Bt tile rows (64x32)

// ---- CDNA5 async global->LDS copy (ASYNCcnt) with sync fallback ----------
#if defined(__has_builtin)
#if __has_builtin(__builtin_amdgcn_global_load_async_to_lds_b128)
#define USE_ASYNC_LDS 1
#endif
#endif
#ifndef USE_ASYNC_LDS
#define USE_ASYNC_LDS 0
#endif

#if USE_ASYNC_LDS
typedef int i4v __attribute__((vector_size(16)));
typedef __attribute__((address_space(1))) i4v* g_i4p;
typedef __attribute__((address_space(3))) i4v* l_i4p;

__device__ __forceinline__ void cp16_async(void* lds_dst, const void* g_src) {
  __builtin_amdgcn_global_load_async_to_lds_b128(
      (g_i4p)g_src, (l_i4p)lds_dst, /*offset=*/0, /*cpol=*/0);
}
__device__ __forceinline__ void wait_async0() {
#if defined(__has_builtin) && __has_builtin(__builtin_amdgcn_s_wait_asynccnt)
  __builtin_amdgcn_s_wait_asynccnt(0);
#else
  asm volatile("s_wait_asynccnt 0x0" ::: "memory");
#endif
}
#endif

// ---------------------------------------------------------------------------
// Tiled WMMA GEMM: C[M x N] = A[M x K] * Bt^T, where Bt is stored (N x K).
// A, Bt are f16 row-major (lda/ldb in halfs). M % 128 == 0, N % 64 == 0,
// K % 32 == 0 (buffers padded). Double-buffered LDS, async staging.
// OM: 0 = f32 C (clipped to m_bound/n_bound), 1 = f16 C, 2 = f16 C stored
// transposed (Ct[n][m], ldc = leading dim of Ct; packed 16B stores).
// ---------------------------------------------------------------------------
template <int OM>
__global__ __launch_bounds__(256) void gemm_wmma(
    const f16_t* __restrict__ A, int lda,
    const f16_t* __restrict__ Bt, int ldb,
    void* __restrict__ Cout, int ldc, int K,
    int m_bound, int n_bound)
{
  __shared__ alignas(16) f16_t lA[2][128 * LDA_S];
  __shared__ alignas(16) f16_t lB[2][64 * LDB_S];   // [n][k]

  const int tid = threadIdx.x;
  const int m0 = blockIdx.y * 128;
  const int n0 = blockIdx.x * 64;
  const int w    = tid >> 5;
  const int lane = tid & 31;
  const int wr = w & 3;        // wave tile row -> rows wr*32
  const int wc = w >> 2;       // wave tile col -> cols wc*32
  const int lm = lane & 15;
  const int hi = lane >> 4;

  auto stage = [&](int buf, int k0) {
    f16_t* la = &lA[buf][0];
    f16_t* lb = &lB[buf][0];
    #pragma unroll
    for (int it = 0; it < 2; ++it) {   // A tile 128x32: 2 x 16B per thread
      int c = tid + it * 256;
      int r = c >> 2, s = c & 3;
      const f16_t* g = &A[(size_t)(m0 + r) * lda + k0 + s * 8];
      f16_t* l = &la[r * LDA_S + s * 8];
#if USE_ASYNC_LDS
      cp16_async(l, g);
#else
      *(uint4*)l = *(const uint4*)g;
#endif
    }
    {                                   // Bt tile 64x32: 1 x 16B per thread
      int n = tid >> 2, s = tid & 3;
      const f16_t* g = &Bt[(size_t)(n0 + n) * ldb + k0 + s * 8];
      f16_t* l = &lb[n * LDB_S + s * 8];
#if USE_ASYNC_LDS
      cp16_async(l, g);
#else
      *(uint4*)l = *(const uint4*)g;
#endif
    }
  };

  v8f acc[2][2] = {};
  stage(0, 0);
  const int nk = K >> 5;

  for (int ki = 0; ki < nk; ++ki) {
    const int cur = ki & 1;
#if USE_ASYNC_LDS
    wait_async0();                 // current buffer's DMA done (this wave)
#endif
    __syncthreads();               // visible to all waves; prev reads retired
    if (ki + 1 < nk) stage(cur ^ 1, (ki + 1) << 5);   // overlap with compute

    union Frag { v16h v; uint4 q[2]; };
    Frag af[2], bf[2];
    const f16_t* la = &lA[cur][0];
    const f16_t* lb = &lB[cur][0];
    #pragma unroll
    for (int f = 0; f < 2; ++f) {
      // A 16x32 fragment: lane lm = row; halves 0..7 <- K=kb..kb+7,
      // halves 8..15 <- K=16+kb..16+kb+7, kb = hi*8
      int m = wr * 32 + f * 16 + lm;
      af[f].q[0] = *(const uint4*)&la[m * LDA_S + hi * 8];
      af[f].q[1] = *(const uint4*)&la[m * LDA_S + 16 + hi * 8];
      // B 32x16 fragment: lane lm = col; halves 0..7 <- K=hi*8..hi*8+7,
      // halves 8..15 <- K=16+hi*8..16+hi*8+7
      int n = wc * 32 + f * 16 + lm;
      bf[f].q[0] = *(const uint4*)&lb[n * LDB_S + hi * 8];
      bf[f].q[1] = *(const uint4*)&lb[n * LDB_S + 16 + hi * 8];
    }
    #pragma unroll
    for (int fr = 0; fr < 2; ++fr)
      #pragma unroll
      for (int fc = 0; fc < 2; ++fc)
        acc[fr][fc] = __builtin_amdgcn_wmma_f32_16x16x32_f16(
            false, af[fr].v, false, bf[fc].v, (short)0, acc[fr][fc], false, false);
  }

  // --- store: C/D layout -> VGPR j = (M = hi*8 + j, N = lm) of each 16x16 ---
  if constexpr (OM == 2) {
    #pragma unroll
    for (int fr = 0; fr < 2; ++fr) {
      #pragma unroll
      for (int fc = 0; fc < 2; ++fc) {
        int cg = n0 + wc * 32 + fc * 16 + lm;
        int rbase = m0 + wr * 32 + fr * 16 + hi * 8;
        union { f16_t h[8]; uint4 q; } pk;
        #pragma unroll
        for (int j = 0; j < 8; ++j) pk.h[j] = (f16_t)acc[fr][fc][j];
        *(uint4*)&((f16_t*)Cout)[(size_t)cg * ldc + rbase] = pk.q;
      }
    }
  } else {
    #pragma unroll
    for (int fr = 0; fr < 2; ++fr) {
      #pragma unroll
      for (int fc = 0; fc < 2; ++fc) {
        int cg = n0 + wc * 32 + fc * 16 + lm;
        if (cg >= n_bound) continue;
        #pragma unroll
        for (int j = 0; j < 8; ++j) {
          int rg = m0 + wr * 32 + fr * 16 + hi * 8 + j;
          if (rg < m_bound) {
            float val = acc[fr][fc][j];
            if constexpr (OM == 1)
              ((f16_t*)Cout)[(size_t)rg * ldc + cg] = (f16_t)val;
            else
              ((float*)Cout)[(size_t)rg * ldc + cg] = val;
          }
        }
      }
    }
  }
}

// ---------------------------------------------------------------------------
// Elementwise / graph kernels
// ---------------------------------------------------------------------------
__global__ void mv_f32_to_f16(const float* __restrict__ in, f16_t* __restrict__ out, size_t n) {
  size_t i = (size_t)blockIdx.x * blockDim.x + threadIdx.x;
  if (i < n) out[i] = (f16_t)in[i];
}

// out[c][r] = (f16) in[r][c]  — pre-transposed f16 weights
__global__ void mv_transpose_convert(const float* __restrict__ in, f16_t* __restrict__ out,
                                     int rows, int cols) {
  int i = blockIdx.x * blockDim.x + threadIdx.x;
  int total = rows * cols;
  if (i >= total) return;
  int r = i / cols, c = i % cols;
  out[(size_t)c * rows + r] = (f16_t)in[i];
}

__global__ void mv_pad_convert(const float* __restrict__ in, f16_t* __restrict__ out,
                               int n_true, int cols, int n_pad) {
  size_t i = (size_t)blockIdx.x * blockDim.x + threadIdx.x;
  size_t total = (size_t)n_pad * cols;
  if (i >= total) return;
  int r = (int)(i / cols);
  out[i] = (r < n_true) ? (f16_t)in[i] : (f16_t)0.0f;
}

__global__ void mv_deg_init(float* __restrict__ deg, int n_true, int n_pad) {
  int i = blockIdx.x * blockDim.x + threadIdx.x;
  if (i < n_pad) deg[i] = (i < n_true) ? 1.0f : 0.0f;  // self-loop weight 1
}

__global__ void mv_deg_scatter(const int* __restrict__ dst, const float* __restrict__ w,
                               float* __restrict__ deg, int me) {
  int i = blockIdx.x * blockDim.x + threadIdx.x;
  if (i < me) atomicAdd(&deg[dst[i]], w[i]);
}

__global__ void mv_deg_rsqrt(const float* __restrict__ deg, float* __restrict__ dis, int n_pad) {
  int i = blockIdx.x * blockDim.x + threadIdx.x;
  if (i < n_pad) dis[i] = (deg[i] > 0.0f) ? rsqrtf(deg[i]) : 0.0f;
}

// agg = dis[i]^2 * h[i]  (self-loop term), zero-inits padded rows
__global__ void mv_selfloop_agg(const float* __restrict__ h, const float* __restrict__ dis,
                                float* __restrict__ agg, int n_true, int n_pad) {
  int idx = blockIdx.x * blockDim.x + threadIdx.x;
  int total = n_pad * 128;
  if (idx >= total) return;
  int i = idx >> 7;
  int f = (idx & 127) << 2;
  float4 o = make_float4(0.f, 0.f, 0.f, 0.f);
  if (i < n_true) {
    float d2 = dis[i] * dis[i];
    float4 hv = *(const float4*)&h[(size_t)i * 512 + f];
    o = make_float4(d2 * hv.x, d2 * hv.y, d2 * hv.z, d2 * hv.w);
  }
  *(float4*)&agg[(size_t)i * 512 + f] = o;
}

// agg[dst] += dis[src]*w*dis[dst] * h[src]   (one thread = one edge x 4 feats)
__global__ void mv_edge_scatter(const int* __restrict__ src, const int* __restrict__ dst,
                                const float* __restrict__ w, const float* __restrict__ dis,
                                const float* __restrict__ h, float* __restrict__ agg, int me) {
  int idx = blockIdx.x * blockDim.x + threadIdx.x;
  int total = me * 128;
  if (idx >= total) return;
  int e = idx >> 7;
  int f = (idx & 127) << 2;
  int s = src[e], d = dst[e];
  float norm = dis[s] * w[e] * dis[d];
  float4 hv = *(const float4*)&h[(size_t)s * 512 + f];
  float* ap = &agg[(size_t)d * 512 + f];
  atomicAdd(ap + 0, norm * hv.x);
  atomicAdd(ap + 1, norm * hv.y);
  atomicAdd(ap + 2, norm * hv.z);
  atomicAdd(ap + 3, norm * hv.w);
}

// x = relu(agg + b); write f32 into emb column block and f16 next-layer input
__global__ void mv_bias_relu_pack(const float* __restrict__ agg, const float* __restrict__ bias,
                                  float* __restrict__ emb, int colofs,
                                  f16_t* __restrict__ xout, int n_true, int n_pad) {
  int idx = blockIdx.x * blockDim.x + threadIdx.x;
  int total = n_pad * 128;
  if (idx >= total) return;
  int i = idx >> 7;
  int f = (idx & 127) << 2;
  float4 v = make_float4(0.f, 0.f, 0.f, 0.f);
  if (i < n_true) {
    float4 a = *(const float4*)&agg[(size_t)i * 512 + f];
    v.x = fmaxf(a.x + bias[f + 0], 0.f);
    v.y = fmaxf(a.y + bias[f + 1], 0.f);
    v.z = fmaxf(a.z + bias[f + 2], 0.f);
    v.w = fmaxf(a.w + bias[f + 3], 0.f);
  }
  *(float4*)&emb[(size_t)i * 2048 + colofs + f] = v;
  union { f16_t h[4]; uint2 u; } pk;
  pk.h[0] = (f16_t)v.x; pk.h[1] = (f16_t)v.y; pk.h[2] = (f16_t)v.z; pk.h[3] = (f16_t)v.w;
  *(uint2*)&xout[(size_t)i * 512 + f] = pk.u;
}

// torch: emb.view(4, E, N).transpose(-1,-2) -> Xv[v][n][e] = emb_flat[v*E*N + e*N + n]
__global__ void mv_permute_xv(const float* __restrict__ emb, f16_t* __restrict__ xv,
                              int n_true, int n_pad) {
  size_t idx = (size_t)blockIdx.x * blockDim.x + threadIdx.x;
  size_t total = (size_t)4 * n_pad * 512;
  if (idx >= total) return;
  int v = (int)(idx / ((size_t)n_pad * 512));
  int rem = (int)(idx % ((size_t)n_pad * 512));
  int n = rem >> 9;
  int e = rem & 511;
  float val = 0.0f;
  if (n < n_true) {
    size_t f = (size_t)v * 512 * n_true + (size_t)e * n_true + n;
    size_t row = f >> 11;        // emb row (4E = 2048 cols)
    size_t col = f & 2047;
    val = emb[row * 2048 + col];
  }
  xv[idx] = (f16_t)val;
}

// row softmax of scores/8 with masking of padded cols/rows, f16 output
__global__ __launch_bounds__(256) void mv_softmax_rows(
    const float* __restrict__ S, f16_t* __restrict__ P, int n_true, int n_pad)
{
  __shared__ float red[256];
  int r = blockIdx.x;
  int tid = threadIdx.x;
  const float* srow = S + (size_t)r * n_pad;
  f16_t* prow = P + (size_t)r * n_pad;
  if (r >= n_true) {
    for (int c = tid; c < n_pad; c += 256) prow[c] = (f16_t)0.0f;
    return;
  }
  float mx = -3.0e38f;
  for (int c = tid; c < n_true; c += 256) mx = fmaxf(mx, srow[c]);
  red[tid] = mx; __syncthreads();
  for (int s = 128; s > 0; s >>= 1) {
    if (tid < s) red[tid] = fmaxf(red[tid], red[tid + s]);
    __syncthreads();
  }
  mx = red[0]; __syncthreads();
  const float scale = 0.125f;  // 1/sqrt(64)
  float sum = 0.f;
  for (int c = tid; c < n_true; c += 256) sum += __expf((srow[c] - mx) * scale);
  red[tid] = sum; __syncthreads();
  for (int s = 128; s > 0; s >>= 1) {
    if (tid < s) red[tid] += red[tid + s];
    __syncthreads();
  }
  float inv = 1.0f / red[0];
  for (int c = tid; c < n_pad; c += 256) {
    float v = (c < n_true) ? __expf((srow[c] - mx) * scale) * inv : 0.0f;
    prow[c] = (f16_t)v;
  }
}

// fuse: y[n][e'] = sum_c fw[c]*Y[c][n][e'] + fb; out[n][e] = y_flat[e*N + n]
__global__ void mv_fuse_out(const float* __restrict__ Y, const float* __restrict__ fw,
                            const float* __restrict__ fb, f16_t* __restrict__ out16,
                            int n_true, int n_pad) {
  int idx = blockIdx.x * blockDim.x + threadIdx.x;
  int total = n_pad * 512;
  if (idx >= total) return;
  int n = idx >> 9;
  int e = idx & 511;
  float val = 0.0f;
  if (n < n_true) {
    size_t g = (size_t)e * n_true + n;
    int n2 = (int)(g >> 9);
    int e2 = (int)(g & 511);
    size_t base = (size_t)n2 * 512 + e2;
    size_t stride = (size_t)n_pad * 512;
    val = fb[0] + fw[0] * Y[base] + fw[1] * Y[base + stride] +
          fw[2] * Y[base + 2 * stride] + fw[3] * Y[base + 3 * stride];
  }
  out16[idx] = (f16_t)val;
}

// ---------------------------------------------------------------------------
// Host orchestration
// ---------------------------------------------------------------------------
static inline int cdiv(int a, int b) { return (a + b - 1) / b; }

static void gemm_launch(hipStream_t s, int om,
                        const f16_t* A, int lda, const f16_t* Bt, int ldb,
                        void* C, int ldc, int M, int N, int K, int mb, int nb) {
  dim3 grid(N / 64, M / 128), blk(256);
  if (om == 0)      gemm_wmma<0><<<grid, blk, 0, s>>>(A, lda, Bt, ldb, C, ldc, K, mb, nb);
  else if (om == 1) gemm_wmma<1><<<grid, blk, 0, s>>>(A, lda, Bt, ldb, C, ldc, K, mb, nb);
  else              gemm_wmma<2><<<grid, blk, 0, s>>>(A, lda, Bt, ldb, C, ldc, K, mb, nb);
}

extern "C" void kernel_launch(void* const* d_in, const int* in_sizes, int n_in,
                              void* d_out, int out_size, void* d_ws, size_t ws_size,
                              hipStream_t stream) {
  (void)in_sizes; (void)n_in; (void)out_size; (void)ws_size;
  const int NM = 1043, ND = 2166;
  const int NMP = 1152, NDP = 2304;          // padded to multiples of 128
  const int EMM = 66752, EDD = 138624;

  const float* mirna_x  = (const float*)d_in[0];
  const float* drug_x   = (const float*)d_in[1];
  const int*   edge_mm1 = (const int*)d_in[2];  const float* attr_mm1 = (const float*)d_in[3];
  const int*   edge_mm2 = (const int*)d_in[4];  const float* attr_mm2 = (const float*)d_in[5];
  const int*   edge_dd1 = (const int*)d_in[6];  const float* attr_dd1 = (const float*)d_in[7];
  const int*   edge_dd2 = (const int*)d_in[8];  const float* attr_dd2 = (const float*)d_in[9];

  // 16 f32 weight matrices (all stored transposed in f16):
  const float* wf32[16] = {
    (const float*)d_in[10], (const float*)d_in[12],   // m1 W1,W2
    (const float*)d_in[14], (const float*)d_in[16],   // m2 W1,W2
    (const float*)d_in[18], (const float*)d_in[20],   // d1 W1,W2
    (const float*)d_in[22], (const float*)d_in[24],   // d2 W1,W2
    (const float*)d_in[26], (const float*)d_in[27], (const float*)d_in[28], (const float*)d_in[29],
    (const float*)d_in[32], (const float*)d_in[33], (const float*)d_in[34], (const float*)d_in[35]
  };
  const float* bm1b1 = (const float*)d_in[11]; const float* bm1b2 = (const float*)d_in[13];
  const float* bm2b1 = (const float*)d_in[15]; const float* bm2b2 = (const float*)d_in[17];
  const float* bd1b1 = (const float*)d_in[19]; const float* bd1b2 = (const float*)d_in[21];
  const float* bd2b1 = (const float*)d_in[23]; const float* bd2b2 = (const float*)d_in[25];
  const float* mir_fw  = (const float*)d_in[30]; const float* mir_fb  = (const float*)d_in[31];
  const float* drug_fw = (const float*)d_in[36]; const float* drug_fb = (const float*)d_in[37];

  // ---- workspace bump allocator ----
  char* wsp = (char*)d_ws; size_t off = 0;
  auto walloc = [&](size_t bytes) -> void* {
    void* p = wsp + off;
    off += (bytes + 255) & ~(size_t)255;
    return p;
  };

  f16_t* Wt16[16];
  for (int i = 0; i < 16; ++i) Wt16[i] = (f16_t*)walloc((size_t)512 * 512 * 2);
  f16_t* Xm16  = (f16_t*)walloc((size_t)NMP * 512 * 2);
  f16_t* Xd16  = (f16_t*)walloc((size_t)NDP * 512 * 2);
  float* deg   = (float*)walloc((size_t)NDP * 4);
  float* dis   = (float*)walloc((size_t)NDP * 4);
  float* hbuf  = (float*)walloc((size_t)NDP * 512 * 4);
  float* agg   = (float*)walloc((size_t)NDP * 512 * 4);
  f16_t* x1f16 = (f16_t*)walloc((size_t)NDP * 512 * 2);
  float* emb   = (float*)walloc((size_t)NDP * 2048 * 4);
  f16_t* Xv16  = (f16_t*)walloc((size_t)4 * NDP * 512 * 2);
  f16_t* Q16   = (f16_t*)walloc((size_t)4 * NDP * 512 * 2);
  f16_t* K16   = (f16_t*)walloc((size_t)4 * NDP * 512 * 2);
  f16_t* Vt16  = (f16_t*)walloc((size_t)4 * NDP * 512 * 2);   // (512 x 4*np) transposed
  float* Sb    = (float*)walloc((size_t)NDP * NDP * 4);
  f16_t* Pb    = (f16_t*)walloc((size_t)NDP * NDP * 2);
  float* Ob    = (float*)walloc((size_t)4 * NDP * 512 * 4);
  f16_t* O16   = (f16_t*)walloc((size_t)4 * NDP * 512 * 2);
  float* Yb    = (float*)walloc((size_t)4 * NDP * 512 * 4);
  f16_t* mir16 = (f16_t*)walloc((size_t)NMP * 512 * 2);
  f16_t* drg16 = (f16_t*)walloc((size_t)NDP * 512 * 2);

  // ---- setup: transpose-convert weights, pad-convert node features ----
  for (int i = 0; i < 16; ++i)
    mv_transpose_convert<<<cdiv(512 * 512, 256), 256, 0, stream>>>(wf32[i], Wt16[i], 512, 512);
  mv_pad_convert<<<cdiv(NMP * 512, 256), 256, 0, stream>>>(mirna_x, Xm16, NM, 512, NMP);
  mv_pad_convert<<<cdiv(NDP * 512, 256), 256, 0, stream>>>(drug_x, Xd16, ND, 512, NDP);

  // ---- one GCN conv layer ----
  auto conv = [&](const f16_t* Xin, const f16_t* Wt, const float* bb,
                  const int* edges, const float* attr, int me,
                  int colofs, int nt, int np) {
    gemm_launch(stream, 0, Xin, 512, Wt, 512, hbuf, 512, np, 512, 512, np, 512);
    mv_selfloop_agg<<<cdiv(np * 128, 256), 256, 0, stream>>>(hbuf, dis, agg, nt, np);
    mv_edge_scatter<<<cdiv(me * 128, 256), 256, 0, stream>>>(edges, edges + me, attr, dis,
                                                             hbuf, agg, me);
    mv_bias_relu_pack<<<cdiv(np * 128, 256), 256, 0, stream>>>(agg, bb, emb, colofs,
                                                               x1f16, nt, np);
  };

  // ---- one GCN encoder (2 layers -> 2 x 512 columns of emb) ----
  auto encoder = [&](const f16_t* X16, const int* edges, const float* attr, int me,
                     const f16_t* Wt1, const float* b1, const f16_t* Wt2, const float* b2,
                     int colbase, int nt, int np) {
    mv_deg_init<<<cdiv(np, 256), 256, 0, stream>>>(deg, nt, np);
    mv_deg_scatter<<<cdiv(me, 256), 256, 0, stream>>>(edges + me, attr, deg, me);
    mv_deg_rsqrt<<<cdiv(np, 256), 256, 0, stream>>>(deg, dis, np);
    conv(X16, Wt1, b1, edges, attr, me, colbase, nt, np);
    conv(x1f16, Wt2, b2, edges, attr, me, colbase + 512, nt, np);
  };

  // ---- MHA over 4 views + fuse ----
  auto attention = [&](const f16_t* Wqt, const f16_t* Wkt, const f16_t* Wvt, const f16_t* Wot,
                       const float* fw, const float* fb, f16_t* out16, int nt, int np) {
    int tot = 4 * np * 512;
    mv_permute_xv<<<cdiv(tot, 256), 256, 0, stream>>>(emb, Xv16, nt, np);
    gemm_launch(stream, 1, Xv16, 512, Wqt, 512, Q16, 512, 4 * np, 512, 512, 4 * np, 512);
    gemm_launch(stream, 1, Xv16, 512, Wkt, 512, K16, 512, 4 * np, 512, 512, 4 * np, 512);
    // V produced pre-transposed: Vt (512 x 4*np), Vt[e][v*np + n]
    gemm_launch(stream, 2, Xv16, 512, Wvt, 512, Vt16, 4 * np, 4 * np, 512, 512, 4 * np, 512);
    for (int v = 0; v < 4; ++v) {
      for (int hh = 0; hh < 8; ++hh) {
        const f16_t* Qs  = Q16  + (size_t)v * np * 512 + hh * 64;
        const f16_t* Ks  = K16  + (size_t)v * np * 512 + hh * 64;
        const f16_t* Vts = Vt16 + (size_t)(hh * 64) * (4 * np) + (size_t)v * np;
        float* Os = Ob + (size_t)v * np * 512 + hh * 64;
        // S = Q_h @ K_h^T  (K slice is (np x 64) = transposed-B storage)
        gemm_launch(stream, 0, Qs, 512, Ks, 512, Sb, np, np, np, 64, np, np);
        mv_softmax_rows<<<np, 256, 0, stream>>>(Sb, Pb, nt, np);
        // O_h = P @ V_h    (V^T slice is (64 x np) = transposed-B storage)
        gemm_launch(stream, 0, Pb, np, Vts, 4 * np, Os, 512, np, 64, np, np, 64);
      }
    }
    mv_f32_to_f16<<<cdiv(tot, 256), 256, 0, stream>>>(Ob, O16, (size_t)tot);
    gemm_launch(stream, 0, O16, 512, Wot, 512, Yb, 512, 4 * np, 512, 512, 4 * np, 512);
    mv_fuse_out<<<cdiv(np * 512, 256), 256, 0, stream>>>(Yb, fw, fb, out16, nt, np);
  };

  // ===== miRNA pipeline =====
  encoder(Xm16, edge_mm1, attr_mm1, EMM, Wt16[0], bm1b1, Wt16[1], bm1b2, 0,    NM, NMP);
  encoder(Xm16, edge_mm2, attr_mm2, EMM, Wt16[2], bm2b1, Wt16[3], bm2b2, 1024, NM, NMP);
  attention(Wt16[8], Wt16[9], Wt16[10], Wt16[11], mir_fw, mir_fb, mir16, NM, NMP);

  // ===== drug pipeline =====
  encoder(Xd16, edge_dd1, attr_dd1, EDD, Wt16[4], bd1b1, Wt16[5], bd1b2, 0,    ND, NDP);
  encoder(Xd16, edge_dd2, attr_dd2, EDD, Wt16[6], bd2b1, Wt16[7], bd2b2, 1024, ND, NDP);
  attention(Wt16[12], Wt16[13], Wt16[14], Wt16[15], drug_fw, drug_fb, drg16, ND, NDP);

  // ===== final: out = mir @ drg^T (drg is (ND x 512) = transposed-B) =====
  gemm_launch(stream, 0, mir16, 512, drg16, 512, d_out, ND,
              NMP, NDP, 512, NM, ND);
}
